// OTTransportLightningModule_55009941127876
// MI455X (gfx1250) — matrix operand order repaired and checked by reading, not compiled, not run on hardware
//
#include <hip/hip_runtime.h>
#include <hip/hip_bf16.h>
#include <cstdint>

typedef __attribute__((ext_vector_type(16))) _Float16 v16h;
typedef __attribute__((ext_vector_type(8)))  float    v8f;
typedef __attribute__((ext_vector_type(4)))  int      v4i;

#define BM 128
#define BN 128
#define BK 32

#define SINK_INV_EPS 10.0f      // 1/0.1
#define SINK_TINY    1.1920929e-7f
#define UNC_FLOOR_C  1e-4f

#if defined(__has_builtin)
#if __has_builtin(__builtin_amdgcn_global_load_async_to_lds_b128)
#define HAVE_ASYNC_LDS 1
#endif
#endif

// 16-byte global->LDS copy: async (ASYNCcnt-tracked) when available.
// Builtin prototype (from hipcc diagnostic): (int4 __device__* src,
// int4 __shared__* dst, Imm offset, Imm cpol) -- pass generic pointers and
// let HIP's builtin-argument AS conversion handle device/shared.
__device__ __forceinline__ void copy16_g2l(const _Float16* g, _Float16* l)
{
#ifdef HAVE_ASYNC_LDS
    __builtin_amdgcn_global_load_async_to_lds_b128(
        (v4i*)(uintptr_t)g, (v4i*)l, 0, 0);
#else
    float4 tmp = *(const float4*)g;
    *(float4*)l = tmp;
#endif
}

__device__ __forceinline__ void wait_async_le4(void)
{
#ifdef HAVE_ASYNC_LDS
    asm volatile("s_wait_asynccnt 0x4" ::: "memory");
#endif
}
__device__ __forceinline__ void wait_async_0(void)
{
#ifdef HAVE_ASYNC_LDS
    asm volatile("s_wait_asynccnt 0x0" ::: "memory");
#endif
}

// ---------------------------------------------------------------------------
// Generic WMMA GEMM (f32 inputs, staged+converted to f16 in LDS).
// mode 0: Out = acc + bias[n]
// mode 1: Out = gelu(acc + bias[n])          (exact erf GELU)
// mode 2: Out += acc + bias[n]               (additive conditioning)
// ---------------------------------------------------------------------------
__global__ __launch_bounds__(256) void wmma_gemm_kernel(
    const float* __restrict__ A, const float* __restrict__ Bm,
    float* __restrict__ Out,
    int M, int N, int K, int mode,
    const float* __restrict__ bias)
{
    __shared__ _Float16 As[BM][BK];
    __shared__ _Float16 Bs[BN][BK];   // stored [n][k]

    const int tid  = threadIdx.x;
    const int lane = tid & 31;
    const int wave = tid >> 5;
    const int bm0  = blockIdx.y * BM;
    const int bn0  = blockIdx.x * BN;
    const int wm0  = (wave >> 2) * 64;   // 0 or 64
    const int wn0  = (wave & 3) * 32;    // 0,32,64,96

    v8f acc[4][2] = {};

    for (int kb = 0; kb < K; kb += BK) {
        // ---- stage A tile (coalesced: lane -> k) ----
        {
            const int kk = tid & 31;
            const int gk = kb + kk;
            for (int m = tid >> 5; m < BM; m += 8) {
                const int gm = bm0 + m;
                float val = (gm < M && gk < K) ? A[(size_t)gm * K + gk] : 0.0f;
                As[m][kk] = (_Float16)val;
            }
        }
        // ---- stage B tile into Bs[n][k] ----
        {
            const int n  = tid & 127;
            const int gn = bn0 + n;
            for (int kk = tid >> 7; kk < BK; kk += 2) {
                const int gk = kb + kk;
                float val = (gn < N && gk < K) ? Bm[(size_t)gk * N + gn] : 0.0f;
                Bs[n][kk] = (_Float16)val;
            }
        }
        __syncthreads();

        // ---- gather fragments ----
        const int mn   = lane & 15;
        const int khiA = (lane & 16) ? 8 : 0;     // A: interleaved K-halves (ISA 16-bit A layout)
        const int khiB = (lane & 16) ? 8 : 0;     // B: interleaved K-halves (same as A)

        v16h afr[4], bfr[2];
        #pragma unroll
        for (int mt = 0; mt < 4; ++mt) {
            const _Float16* pa = &As[wm0 + mt * 16 + mn][0];
            #pragma unroll
            for (int g = 0; g < 4; ++g) {
                const int ks = khiA + g * 4 + ((g & 2) ? 8 : 0);
                #pragma unroll
                for (int e = 0; e < 4; ++e) afr[mt][g * 4 + e] = pa[ks + e];
            }
        }
        #pragma unroll
        for (int nt = 0; nt < 2; ++nt) {
            const _Float16* pb = &Bs[wn0 + nt * 16 + mn][0];
            #pragma unroll
            for (int h = 0; h < 16; ++h) bfr[nt][h] = pb[khiB + h + ((h & 8) ? 8 : 0)];
        }

        #pragma unroll
        for (int mt = 0; mt < 4; ++mt)
            #pragma unroll
            for (int nt = 0; nt < 2; ++nt)
                acc[mt][nt] = __builtin_amdgcn_wmma_f32_16x16x32_f16(
                    false, afr[mt], false, bfr[nt],
                    (short)0, acc[mt][nt], false, false);
        __syncthreads();
    }

    // ---- epilogue ----
    const int rbase = (lane & 16) ? 8 : 0;
    const int col   = lane & 15;
    #pragma unroll
    for (int mt = 0; mt < 4; ++mt) {
        #pragma unroll
        for (int nt = 0; nt < 2; ++nt) {
            const int gn = bn0 + wn0 + nt * 16 + col;
            #pragma unroll
            for (int r = 0; r < 8; ++r) {
                const int gm = bm0 + wm0 + mt * 16 + rbase + r;
                if (gm < M && gn < N) {
                    const float va   = acc[mt][nt][r];
                    const size_t idx = (size_t)gm * N + gn;
                    if (mode == 0) {
                        Out[idx] = va + bias[gn];
                    } else if (mode == 1) {
                        const float x = va + bias[gn];
                        Out[idx] = 0.5f * x * (1.0f + erff(x * 0.70710678118654752f));
                    } else {
                        Out[idx] += va + bias[gn];
                    }
                }
            }
        }
    }
}

// ---------------------------------------------------------------------------
// Specialized cost-matrix WMMA GEMM: A16[M,K] @ B16[N,K]^T, f16 operands,
// async double-buffered global->LDS staging, fused Sinkhorn epilogue:
//   cost = max(rn[m] + cn[n] - 2*acc, 0);  Cost=f32;  Km=bf16(exp(-cost/eps))
// Requires: M,N multiples of 128; K multiple of 8.
// ---------------------------------------------------------------------------
__global__ __launch_bounds__(256) void wmma_cost_kernel(
    const _Float16* __restrict__ A16, const _Float16* __restrict__ B16,
    int M, int N, int K,
    const float* __restrict__ rn, const float* __restrict__ cn,
    float* __restrict__ cost_out, __hip_bfloat16* __restrict__ km_out)
{
    __shared__ _Float16 As[2][BM][BK];
    __shared__ _Float16 Bs[2][BN][BK];

    const int tid  = threadIdx.x;
    const int lane = tid & 31;
    const int wave = tid >> 5;
    const int bm0  = blockIdx.y * BM;
    const int bn0  = blockIdx.x * BN;
    const int wm0  = (wave >> 2) * 64;
    const int wn0  = (wave & 3) * 32;

    // Per-wave staging: 2 async b128 instructions for A + 2 for B per tile.
    // Each instruction: 32 lanes x 16B; 4 lanes per 64B row, 8 rows/instr.
    const int kkg = (lane & 3) * 8;             // K-granule start within tile
    auto stage = [&](int buf, int kb) {
        const int  gk   = kb + kkg;
        const bool kval = (gk + 8) <= K;        // granule fully valid (K%8==0)
        #pragma unroll
        for (int i = 0; i < 2; ++i) {
            const int row = wave * 16 + i * 8 + (lane >> 2);
            _Float16* la = &As[buf][row][kkg];
            _Float16* lb = &Bs[buf][row][kkg];
            if (kval) {
                copy16_g2l(A16 + (size_t)(bm0 + row) * K + gk, la);
                copy16_g2l(B16 + (size_t)(bn0 + row) * K + gk, lb);
            } else {
                const float4 zz = {0.0f, 0.0f, 0.0f, 0.0f};
                *(float4*)la = zz;
                *(float4*)lb = zz;
            }
        }
    };

    v8f acc[4][2] = {};
    const int nkt = (K + BK - 1) / BK;
    int buf = 0;
    stage(0, 0);

    for (int t = 0; t < nkt; ++t) {
        const bool hasNext = (t + 1) < nkt;
        if (hasNext) stage(buf ^ 1, (t + 1) * BK);
        if (hasNext) wait_async_le4(); else wait_async_0();
        __syncthreads();

        const int mn   = lane & 15;
        const int khiA = (lane & 16) ? 8 : 0;
        const int khiB = (lane & 16) ? 8 : 0;

        v16h afr[4], bfr[2];
        #pragma unroll
        for (int mt = 0; mt < 4; ++mt) {
            const _Float16* pa = &As[buf][wm0 + mt * 16 + mn][0];
            #pragma unroll
            for (int g = 0; g < 4; ++g) {
                const int ks = khiA + g * 4 + ((g & 2) ? 8 : 0);
                #pragma unroll
                for (int e = 0; e < 4; ++e) afr[mt][g * 4 + e] = pa[ks + e];
            }
        }
        #pragma unroll
        for (int nt = 0; nt < 2; ++nt) {
            const _Float16* pb = &Bs[buf][wn0 + nt * 16 + mn][0];
            #pragma unroll
            for (int h = 0; h < 16; ++h) bfr[nt][h] = pb[khiB + h + ((h & 8) ? 8 : 0)];
        }

        #pragma unroll
        for (int mt = 0; mt < 4; ++mt)
            #pragma unroll
            for (int nt = 0; nt < 2; ++nt)
                acc[mt][nt] = __builtin_amdgcn_wmma_f32_16x16x32_f16(
                    false, afr[mt], false, bfr[nt],
                    (short)0, acc[mt][nt], false, false);

        __syncthreads();   // all waves done reading buf before it is refilled
        buf ^= 1;
    }

    const int rbase = (lane & 16) ? 8 : 0;
    const int col   = lane & 15;
    #pragma unroll
    for (int mt = 0; mt < 4; ++mt) {
        #pragma unroll
        for (int nt = 0; nt < 2; ++nt) {
            const int gn = bn0 + wn0 + nt * 16 + col;
            #pragma unroll
            for (int r = 0; r < 8; ++r) {
                const int gm = bm0 + wm0 + mt * 16 + rbase + r;
                if (gm < M && gn < N) {
                    const size_t idx = (size_t)gm * N + gn;
                    const float  cst = fmaxf(rn[gm] + cn[gn] - 2.0f * acc[mt][nt][r], 0.0f);
                    cost_out[idx] = cst;
                    km_out[idx]   = __float2bfloat16(__builtin_expf(-cst * SINK_INV_EPS));
                }
            }
        }
    }
}

// ---------------------------------------------------------------------------
// helpers
// ---------------------------------------------------------------------------
__device__ inline float block_reduce_sum(float s)
{
    __shared__ float red[8];
    #pragma unroll
    for (int off = 16; off > 0; off >>= 1) s += __shfl_xor(s, off, 32);
    if ((threadIdx.x & 31) == 0) red[threadIdx.x >> 5] = s;
    __syncthreads();
    if (threadIdx.x == 0) {
        s = red[0];
        #pragma unroll
        for (int i = 1; i < 8; ++i) s += red[i];
    }
    return s;
}

__global__ __launch_bounds__(256) void f32_to_f16_kernel(
    const float* __restrict__ in, _Float16* __restrict__ out, size_t n)
{
    size_t i = (size_t)blockIdx.x * 256 + threadIdx.x;
    const size_t stride = (size_t)gridDim.x * 256;
    for (; i < n; i += stride) out[i] = (_Float16)in[i];
}

__global__ __launch_bounds__(256) void row_sqnorm_kernel(
    const float* __restrict__ X, float* __restrict__ out, int cols)
{
    const int row = blockIdx.x;
    const float* xr = X + (size_t)row * cols;
    float s = 0.0f;
    for (int j = threadIdx.x; j < cols; j += 256) { const float x = xr[j]; s += x * x; }
    s = block_reduce_sum(s);
    if (threadIdx.x == 0) out[row] = s;
}

__global__ __launch_bounds__(256) void softplus_floor_kernel(float* __restrict__ v, size_t n)
{
    size_t i = (size_t)blockIdx.x * 256 + threadIdx.x;
    const size_t stride = (size_t)gridDim.x * 256;
    for (; i < n; i += stride) {
        const float x  = v[i];
        const float sp = (x > 20.0f) ? x : log1pf(__builtin_expf(x));
        v[i] = sp + UNC_FLOOR_C;
    }
}

__global__ void fill_f32_kernel(float* __restrict__ p, float val, int n)
{
    const int i = blockIdx.x * 256 + threadIdx.x;
    if (i < n) p[i] = val;
}

__global__ void zero_acc_kernel(double* __restrict__ a)
{
    if (threadIdx.x < 2) a[threadIdx.x] = 0.0;
}

// u[i] = a / max(sum_j K[i,j] * v[j], tiny)   -- one block per row, L2-resident K
__global__ __launch_bounds__(256) void sink_rows_kernel(
    const __hip_bfloat16* __restrict__ Km, const float* __restrict__ v,
    float* __restrict__ u, int n, float a)
{
    const int row = blockIdx.x;
    const __hip_bfloat16* kr = Km + (size_t)row * n;
    float s = 0.0f;
    for (int j = threadIdx.x; j < n; j += 256)
        s += __bfloat162float(kr[j]) * v[j];
    s = block_reduce_sum(s);
    if (threadIdx.x == 0) u[row] = a / fmaxf(s, SINK_TINY);
}

// partial column sums: vtmp[col] += sum over a 128-row chunk of K[row,col]*u[row]
__global__ __launch_bounds__(256) void sink_cols_partial_kernel(
    const __hip_bfloat16* __restrict__ Km, const float* __restrict__ u,
    float* __restrict__ vtmp, int n, int rows_per_block)
{
    const int col = blockIdx.x * 256 + threadIdx.x;
    const int r0  = blockIdx.y * rows_per_block;
    float s = 0.0f;
    for (int i = 0; i < rows_per_block; ++i) {
        const int row = r0 + i;
        s += __bfloat162float(Km[(size_t)row * n + col]) * u[row];
    }
    atomicAdd(&vtmp[col], s);
}

__global__ void sink_cols_final_kernel(const float* __restrict__ vtmp,
                                       float* __restrict__ v, int n, float b)
{
    const int i = blockIdx.x * 256 + threadIdx.x;
    if (i < n) v[i] = b / fmaxf(vtmp[i], SINK_TINY);
}

__global__ __launch_bounds__(256) void nll_partial_kernel(
    const float* __restrict__ mean, const float* __restrict__ var,
    const float* __restrict__ tgt, double* __restrict__ acc, size_t n)
{
    size_t i = (size_t)blockIdx.x * 256 + threadIdx.x;
    const size_t stride = (size_t)gridDim.x * 256;
    float s = 0.0f;
    for (; i < n; i += stride) {
        const float d  = tgt[i] - mean[i];
        const float vv = var[i];
        s += d * d / vv + __builtin_logf(vv);
    }
    s = block_reduce_sum(s);
    if (threadIdx.x == 0) atomicAdd(acc + 0, (double)s);
}

__global__ __launch_bounds__(256) void plan_cost_partial_kernel(
    const float* __restrict__ Cost, const float* __restrict__ u,
    const float* __restrict__ v, double* __restrict__ acc,
    int n, int rows_per_block)
{
    const int col = blockIdx.x * 256 + threadIdx.x;
    const int r0  = blockIdx.y * rows_per_block;
    const float vc = v[col];
    float s = 0.0f;
    for (int i = 0; i < rows_per_block; ++i) {
        const int row = r0 + i;
        const float c = Cost[(size_t)row * n + col];
        s += u[row] * __builtin_expf(-c * SINK_INV_EPS) * vc * c;
    }
    s = block_reduce_sum(s);
    if (threadIdx.x == 0) atomicAdd(acc + 1, (double)s);
}

__global__ void finalize_kernel(const double* __restrict__ acc,
                                float* __restrict__ out, double inv_bd)
{
    if (threadIdx.x == 0 && blockIdx.x == 0)
        out[0] = (float)(0.5 * acc[0] * inv_bd + 0.1 * acc[1]);
}

// ---------------------------------------------------------------------------
extern "C" void kernel_launch(void* const* d_in, const int* in_sizes, int n_in,
                              void* d_out, int out_size, void* d_ws, size_t ws_size,
                              hipStream_t stream)
{
    (void)in_sizes; (void)n_in; (void)out_size; (void)ws_size;
    const int B = 8192, D = 2000, S = 512, C = 128, L = 64, H = 128;

    const float* src = (const float*)d_in[0];
    const float* tgt = (const float*)d_in[1];
    const float* seq = (const float*)d_in[2];
    const float* cel = (const float*)d_in[3];
    const float* We1 = (const float*)d_in[4];  const float* be1 = (const float*)d_in[5];
    const float* We2 = (const float*)d_in[6];  const float* be2 = (const float*)d_in[7];
    const float* Wsm = (const float*)d_in[8];  const float* bsm = (const float*)d_in[9];
    const float* Wcm = (const float*)d_in[10]; const float* bcm = (const float*)d_in[11];
    const float* Wm1 = (const float*)d_in[12]; const float* bm1 = (const float*)d_in[13];
    const float* Wm2 = (const float*)d_in[14]; const float* bm2 = (const float*)d_in[15];
    const float* Wv1 = (const float*)d_in[16]; const float* bv1 = (const float*)d_in[17];
    const float* Wv2 = (const float*)d_in[18]; const float* bv2 = (const float*)d_in[19];
    float* out = (float*)d_out;

    // ---- workspace carve ----
    char* p = (char*)d_ws;
    auto carve = [&](size_t bytes) -> void* {
        void* r = (void*)p;
        p += (bytes + 255) & ~(size_t)255;
        return r;
    };
    float* H1    = (float*)carve((size_t)B * H * 4);
    float* z     = (float*)carve((size_t)B * L * 4);
    float* Hm    = (float*)carve((size_t)B * H * 4);
    float* Mean  = (float*)carve((size_t)B * D * 4);
    float* Hv    = (float*)carve((size_t)B * H * 4);
    float* Var   = (float*)carve((size_t)B * D * 4);
    float* mnorm = (float*)carve((size_t)B * 4);
    float* tnorm = (float*)carve((size_t)B * 4);
    float* Cost  = (float*)carve((size_t)B * B * 4);
    __hip_bfloat16* Km = (__hip_bfloat16*)carve((size_t)B * B * 2);
    _Float16* Mean16 = (_Float16*)carve((size_t)B * D * 2);
    _Float16* Tgt16  = (_Float16*)carve((size_t)B * D * 2);
    float* u    = (float*)carve((size_t)B * 4);
    float* v    = (float*)carve((size_t)B * 4);
    float* vtmp = (float*)carve((size_t)B * 4);
    double* acc = (double*)carve(2 * sizeof(double));

    const dim3 blk(256);
    auto gemm = [&](const float* A, const float* Bm, float* Out,
                    int M, int N, int K, int mode, const float* bias) {
        dim3 grid((N + BN - 1) / BN, (M + BM - 1) / BM);
        wmma_gemm_kernel<<<grid, blk, 0, stream>>>(A, Bm, Out, M, N, K, mode, bias);
    };

    // ---- encoder: H1 = gelu(src @ We1 + be1); z = H1 @ We2 + be2 + conds ----
    gemm(src, We1, H1, B, H, D, 1, be1);
    gemm(H1,  We2, z,  B, L, H, 0, be2);
    gemm(seq, Wsm, z,  B, L, S, 2, bsm);
    gemm(cel, Wcm, z,  B, L, C, 2, bcm);

    // ---- decoders ----
    gemm(z,  Wm1, Hm,   B, H, L, 1, bm1);
    gemm(Hm, Wm2, Mean, B, D, H, 0, bm2);
    gemm(z,  Wv1, Hv,   B, H, L, 1, bv1);
    gemm(Hv, Wv2, Var,  B, D, H, 0, bv2);
    softplus_floor_kernel<<<2048, blk, 0, stream>>>(Var, (size_t)B * D);

    // ---- cost matrix + bf16 exp-kernel (f16 operands, async-staged WMMA) ----
    row_sqnorm_kernel<<<B, blk, 0, stream>>>(Mean, mnorm, D);
    row_sqnorm_kernel<<<B, blk, 0, stream>>>(tgt,  tnorm, D);
    f32_to_f16_kernel<<<2048, blk, 0, stream>>>(Mean, Mean16, (size_t)B * D);
    f32_to_f16_kernel<<<2048, blk, 0, stream>>>(tgt,  Tgt16,  (size_t)B * D);
    {
        dim3 grid(B / BN, B / BM);
        wmma_cost_kernel<<<grid, blk, 0, stream>>>(Mean16, Tgt16, B, B, D,
                                                   mnorm, tnorm, Cost, Km);
    }

    // ---- Sinkhorn iterations (K bf16 = 134 MB, L2-resident) ----
    const float ab = 1.0f / (float)B;
    fill_f32_kernel<<<B / 256, blk, 0, stream>>>(u, 1.0f, B);
    fill_f32_kernel<<<B / 256, blk, 0, stream>>>(v, 1.0f, B);
    for (int it = 0; it < 30; ++it) {
        sink_rows_kernel<<<B, blk, 0, stream>>>(Km, v, u, B, ab);
        fill_f32_kernel<<<B / 256, blk, 0, stream>>>(vtmp, 0.0f, B);
        sink_cols_partial_kernel<<<dim3(B / 256, B / 128), blk, 0, stream>>>(Km, u, vtmp, B, 128);
        sink_cols_final_kernel<<<B / 256, blk, 0, stream>>>(vtmp, v, B, ab);
    }

    // ---- loss reductions ----
    zero_acc_kernel<<<1, 32, 0, stream>>>(acc);
    nll_partial_kernel<<<4096, blk, 0, stream>>>(Mean, Var, tgt, acc, (size_t)B * D);
    plan_cost_partial_kernel<<<dim3(B / 256, B / 128), blk, 0, stream>>>(Cost, u, v, acc, B, 128);
    finalize_kernel<<<1, 32, 0, stream>>>(acc, out, 1.0 / ((double)B * (double)D));
}
